// NerfModel_9929964388707
// MI455X (gfx1250) — matrix operand order repaired and checked by reading, not compiled, not run on hardware
//
#include <hip/hip_runtime.h>
#include <hip/hip_bf16.h>

// ---------------------------------------------------------------------------
// NeRF forward for MI455X (gfx1250, wave32, WMMA).
// - All GEMMs: v_wmma_f32_16x16x32_f16, f32 accumulation.
// - Weights converted once to f16 transposed/padded; each wave hoists its two
//   16-col weight tiles into VGPRs once per layer (weights read once/block).
// - Activations ping-pong in LDS (stride padded to kill bank conflicts);
//   184 KB/workgroup, allowed by CDNA5's 320KB WGP LDS.
// - Input encodings staged to LDS via the CDNA5 async path (ASYNCcnt).
// ---------------------------------------------------------------------------

typedef __attribute__((ext_vector_type(16))) _Float16 v16h;
typedef __attribute__((ext_vector_type(8)))  _Float16 v8h;
typedef __attribute__((ext_vector_type(8)))  float    v8f;
typedef __attribute__((ext_vector_type(4)))  int      v4i;

#define RAYS 1024
#define SC   64
#define SF   192
#define TM   128       // points per workgroup in the fused MLP kernel
#define LDST 328       // LDS activation row stride (halves); 328*2=656B, 656/16
                       // odd -> 16 rows hit 16 distinct 16B bank slots.

#if defined(__has_builtin)
#if __has_builtin(__builtin_amdgcn_global_load_async_to_lds_b128) && \
    __has_builtin(__builtin_amdgcn_s_wait_asynccnt)
#define HAVE_ASYNC_LDS 1
#endif
#endif
#ifndef HAVE_ASYNC_LDS
#define HAVE_ASYNC_LDS 0
#endif

// ---------------- WMMA operand loaders (per CDNA5 ISA 7.12.2 layouts) -------

// A operand: 16x32 f16 tile from LDS. lane&15 = M row, lane>>4 selects K-half.
// Halves 0..7  hold K = kBase + khi*8 + {0..7}
// Halves 8..15 hold K = kBase + 16 + khi*8 + {0..7}
__device__ inline v16h load_a(const _Float16* __restrict__ act, int stride,
                              int rowBase, int kBase, int lane) {
  int m   = rowBase + (lane & 15);
  int khi = (lane >> 4) & 1;
  const _Float16* p = act + m * stride + kBase + khi * 8;
  v8h lo = *(const v8h*)(p);
  v8h hi = *(const v8h*)(p + 16);
  v16h a;
#pragma unroll
  for (int i = 0; i < 8; ++i) { a[i] = lo[i]; a[i + 8] = hi[i]; }
  return a;
}

// B operand: 32x16 f16 tile from transposed weights Wt[n][k] (k contiguous).
// lane&15 = N col; halves 0..7 hold K = kBase + khi*8 + {0..7}, 8..15 hold K = kBase + 16 + khi*8 + {0..7}.
template <int KP>
__device__ inline v16h load_b(const _Float16* __restrict__ Wt,
                              int colBase, int kBase, int lane) {
  int n    = colBase + (lane & 15);
  int koff = kBase + ((lane >> 4) & 1) * 8;
  const _Float16* p = Wt + (long)n * KP + koff;
  v8h lo = *(const v8h*)(p);
  v8h hi = *(const v8h*)(p + 16);
  v16h b;
#pragma unroll
  for (int i = 0; i < 8; ++i) { b[i] = lo[i]; b[i + 8] = hi[i]; }
  return b;
}

// ---------------- fused GEMM layers ----------------------------------------

// act[TM][actStride] (f16, LDS) x Wt[NCT*16][KP] -> out[TM][LDST] (f16, LDS)
// 8 waves; wave w owns column tiles {w*CPW .. w*CPW+CPW-1}; its weight tiles
// are hoisted into VGPRs once, then reused across all 8 row tiles with CPW
// independent accumulator chains per row tile.
template <int KP, int NCT, bool RELU>
__device__ inline void gemm_lds(const _Float16* __restrict__ act, int actStride,
                                const _Float16* __restrict__ Wt,
                                const float* __restrict__ bias,
                                _Float16* __restrict__ out,
                                int wave, int lane) {
  constexpr int KT  = KP / 32;   // K steps
  constexpr int CPW = NCT / 8;   // column tiles per wave (1 or 2)
  __builtin_prefetch(Wt + (long)(wave * CPW * 16) * KP, 0, 1);
  v16h B[CPW][KT];
#pragma unroll
  for (int c = 0; c < CPW; ++c)
#pragma unroll
    for (int kk = 0; kk < KT; ++kk)
      B[c][kk] = load_b<KP>(Wt, (wave * CPW + c) * 16, kk * 32, lane);

#pragma unroll 1
  for (int rt = 0; rt < 8; ++rt) {
    v8f acc[CPW];
    v8f zero = {};
#pragma unroll
    for (int c = 0; c < CPW; ++c) acc[c] = zero;
#pragma unroll
    for (int kk = 0; kk < KT; ++kk) {
      v16h a = load_a(act, actStride, rt * 16, kk * 32, lane);
#pragma unroll
      for (int c = 0; c < CPW; ++c)
        acc[c] = __builtin_amdgcn_wmma_f32_16x16x32_f16(
            false, a, false, B[c][kk], (short)0, acc[c], false, false);
    }
    int m0 = rt * 16 + ((lane >> 4) & 1) * 8;
#pragma unroll
    for (int c = 0; c < CPW; ++c) {
      int n = (wave * CPW + c) * 16 + (lane & 15);
      float bv = bias[n];
#pragma unroll
      for (int r = 0; r < 8; ++r) {
        float v = acc[c][r] + bv;
        if (RELU) v = v > 0.0f ? v : 0.0f;
        out[(m0 + r) * LDST + n] = (_Float16)v;
      }
    }
  }
}

// head: act x Wt[16][KP] -> global f32, only first Nreal columns valid.
// wave w computes row tile w (single column tile).
template <int KP>
__device__ inline void gemm_head(const _Float16* __restrict__ act, int actStride,
                                 const _Float16* __restrict__ Wt,
                                 const float* __restrict__ bias, int Nreal,
                                 float* __restrict__ outG, int outLd, long p0,
                                 int wave, int lane) {
  const int rt = wave;
  v8f acc = {};
#pragma unroll
  for (int kk = 0; kk < KP; kk += 32) {
    v16h a = load_a(act, actStride, rt * 16, kk, lane);
    v16h b = load_b<KP>(Wt, 0, kk, lane);
    acc = __builtin_amdgcn_wmma_f32_16x16x32_f16(
        false, a, false, b, (short)0, acc, false, false);
  }
  int n = lane & 15;
  if (n < Nreal) {
    float bv = bias[n];
    int m0 = rt * 16 + ((lane >> 4) & 1) * 8;
#pragma unroll
    for (int r = 0; r < 8; ++r)
      outG[(p0 + m0 + r) * outLd + n] = acc[r] + bv;
  }
}

// ---------------- fused MLP kernel -----------------------------------------

struct MlpW {
  const _Float16 *w0, *w1, *w2, *w3, *w4, *w5, *w6, *w7;
  const _Float16 *wsig, *wbneck, *wcond, *wrgb;
  const float *b0, *b1, *b2, *b3, *b4, *b5, *b6, *b7;
  const float *bsig, *bbneck, *bcond, *brgb;
};

__global__ __launch_bounds__(256)
void nerf_mlp(const _Float16* __restrict__ enc,    // [Npts][64] f16 (posenc, padded)
              const _Float16* __restrict__ vdenc,  // [RAYS][32] f16 (padded)
              MlpW W,
              float* __restrict__ sigOut,          // [Npts]
              float* __restrict__ rgbOut,          // [Npts][3]
              int S) {
  extern __shared__ _Float16 lds[];
  _Float16* bufA = lds;                 // TM*LDST
  _Float16* bufB = lds + TM * LDST;     // TM*LDST
  _Float16* inp  = lds + 2 * TM * LDST; // TM*64
  int tid = threadIdx.x, wave = tid >> 5, lane = tid & 31;
  long p0 = (long)blockIdx.x * TM;

  // stage input encodings into LDS (CDNA5 async-to-LDS path, ASYNCcnt)
#if HAVE_ASYNC_LDS
  typedef __attribute__((address_space(1))) v4i gvec;
  typedef __attribute__((address_space(3))) v4i lvec;
  for (int i = tid; i < TM * 64 / 8; i += 256) {
    __builtin_amdgcn_global_load_async_to_lds_b128(
        (gvec*)(enc + p0 * 64 + (long)i * 8),
        (lvec*)(inp + i * 8), 0, 0);
  }
  __builtin_amdgcn_s_wait_asynccnt(0);
#else
  {
    const v16h* s = (const v16h*)(enc + p0 * 64);
    v16h* d = (v16h*)inp;
    for (int i = tid; i < TM * 64 / 16; i += 256) d[i] = s[i];
  }
#endif
  __syncthreads();

  gemm_lds< 64, 16, true >(inp,  64,  W.w0, W.b0, bufA, wave, lane); __syncthreads();
  gemm_lds<256, 16, true >(bufA, LDST, W.w1, W.b1, bufB, wave, lane); __syncthreads();
  gemm_lds<256, 16, true >(bufB, LDST, W.w2, W.b2, bufA, wave, lane); __syncthreads();
  gemm_lds<256, 16, true >(bufA, LDST, W.w3, W.b3, bufB, wave, lane); __syncthreads();
  gemm_lds<256, 16, true >(bufB, LDST, W.w4, W.b4, bufA, wave, lane); __syncthreads();
  // skip concat: append input encoding (63 + zero pad) at cols 256..319
  for (int i = tid; i < TM * 64; i += 256) {
    int m = i >> 6, c = i & 63;
    bufA[m * LDST + 256 + c] = inp[m * 64 + c];
  }
  __syncthreads();
  gemm_lds<320, 16, true >(bufA, LDST, W.w5, W.b5, bufB, wave, lane); __syncthreads();
  gemm_lds<256, 16, true >(bufB, LDST, W.w6, W.b6, bufA, wave, lane); __syncthreads();
  gemm_lds<256, 16, true >(bufA, LDST, W.w7, W.b7, bufB, wave, lane); __syncthreads();

  // sigma head (raw, relu applied in render) + bottleneck (no relu)
  gemm_head<256>(bufB, LDST, W.wsig, W.bsig, 1, sigOut, 1, p0, wave, lane);
  gemm_lds<256, 16, false>(bufB, LDST, W.wbneck, W.bbneck, bufA, wave, lane);
  // append per-ray view encoding (27 + zero pad) at cols 256..287
  for (int i = tid; i < TM * 32; i += 256) {
    int m = i >> 5, c = i & 31;
    long ray = (p0 + m) / S;
    bufA[m * LDST + 256 + c] = vdenc[ray * 32 + c];
  }
  __syncthreads();
  gemm_lds<288, 8, true >(bufA, LDST, W.wcond, W.bcond, bufB, wave, lane);
  __syncthreads();
  gemm_head<128>(bufB, LDST, W.wrgb, W.brgb, 3, rgbOut, 3, p0, wave, lane);
}

// ---------------- small kernels --------------------------------------------

// fp32 W[fi][fo] -> f16 Wt[Np][Kp] transposed + zero padded
__global__ void convert_wt(const float* __restrict__ W, int fi, int fo,
                           int Kp, int Np, _Float16* __restrict__ dst) {
  int idx = blockIdx.x * blockDim.x + threadIdx.x;
  if (idx >= Np * Kp) return;
  int n = idx / Kp, k = idx - n * Kp;
  float v = (k < fi && n < fo) ? W[(long)k * fo + n] : 0.0f;
  dst[idx] = (_Float16)v;
}

__device__ inline void posenc64(float cx, float cy, float cz,
                                _Float16* __restrict__ e) {
  e[0] = (_Float16)cx; e[1] = (_Float16)cy; e[2] = (_Float16)cz;
  float c3[3] = {cx, cy, cz};
#pragma unroll
  for (int d = 0; d < 10; ++d) {
    float sc = (float)(1 << d);
#pragma unroll
    for (int c = 0; c < 3; ++c) {
      float v = c3[c] * sc;
      e[3 + d * 3 + c]  = (_Float16)__sinf(v);
      e[33 + d * 3 + c] = (_Float16)__cosf(v);
    }
  }
  e[63] = (_Float16)0.0f;
}

__global__ void coarse_enc(const float* __restrict__ o, const float* __restrict__ d,
                           _Float16* __restrict__ enc, float* __restrict__ zv) {
  int idx = blockIdx.x * blockDim.x + threadIdx.x;
  if (idx >= RAYS * SC) return;
  int ray = idx >> 6, s = idx & 63;
  float t = (float)s * (1.0f / 63.0f);
  float z = 2.0f * (1.0f - t) + 6.0f * t;
  zv[idx] = z;
  float cx = o[ray * 3 + 0] + z * d[ray * 3 + 0];
  float cy = o[ray * 3 + 1] + z * d[ray * 3 + 1];
  float cz = o[ray * 3 + 2] + z * d[ray * 3 + 2];
  posenc64(cx, cy, cz, enc + (long)idx * 64);
}

__global__ void fine_enc(const float* __restrict__ o, const float* __restrict__ d,
                         const float* __restrict__ zv, _Float16* __restrict__ enc) {
  int idx = blockIdx.x * blockDim.x + threadIdx.x;
  if (idx >= RAYS * SF) return;
  int ray = idx / SF;
  float z = zv[idx];
  float cx = o[ray * 3 + 0] + z * d[ray * 3 + 0];
  float cy = o[ray * 3 + 1] + z * d[ray * 3 + 1];
  float cz = o[ray * 3 + 2] + z * d[ray * 3 + 2];
  posenc64(cx, cy, cz, enc + (long)idx * 64);
}

__global__ void vd_enc(const float* __restrict__ vd, _Float16* __restrict__ out) {
  int r = blockIdx.x * blockDim.x + threadIdx.x;
  if (r >= RAYS) return;
  _Float16* e = out + r * 32;
  float c3[3] = {vd[r * 3], vd[r * 3 + 1], vd[r * 3 + 2]};
  e[0] = (_Float16)c3[0]; e[1] = (_Float16)c3[1]; e[2] = (_Float16)c3[2];
#pragma unroll
  for (int d = 0; d < 4; ++d) {
    float sc = (float)(1 << d);
#pragma unroll
    for (int c = 0; c < 3; ++c) {
      float v = c3[c] * sc;
      e[3 + d * 3 + c]  = (_Float16)__sinf(v);
      e[15 + d * 3 + c] = (_Float16)__cosf(v);
    }
  }
  for (int c = 27; c < 32; ++c) e[c] = (_Float16)0.0f;
}

__global__ void render(const float* __restrict__ rgbRaw, const float* __restrict__ sigRaw,
                       const float* __restrict__ zv, const float* __restrict__ dirs,
                       float* __restrict__ oRgb, float* __restrict__ oDisp,
                       float* __restrict__ oAcc, float* __restrict__ wOut, int S) {
  int r = blockIdx.x * blockDim.x + threadIdx.x;
  if (r >= RAYS) return;
  float dx = dirs[r * 3], dy = dirs[r * 3 + 1], dz = dirs[r * 3 + 2];
  float dn = sqrtf(dx * dx + dy * dy + dz * dz);
  float T = 1.0f, acc = 0.0f, depth = 0.0f;
  float cr = 0.0f, cg = 0.0f, cb = 0.0f;
  const float eps = 1e-10f;
  for (int s = 0; s < S; ++s) {
    long i = (long)r * S + s;
    float z = zv[i];
    float dist = (s + 1 < S) ? (zv[i + 1] - z) * dn : 1e10f * dn;
    float sg = sigRaw[i]; sg = sg > 0.0f ? sg : 0.0f;
    float alpha = 1.0f - __expf(-sg * dist);
    float w = alpha * T;
    cr += w * (1.0f / (1.0f + __expf(-rgbRaw[i * 3 + 0])));
    cg += w * (1.0f / (1.0f + __expf(-rgbRaw[i * 3 + 1])));
    cb += w * (1.0f / (1.0f + __expf(-rgbRaw[i * 3 + 2])));
    depth += w * z; acc += w;
    if (wOut) wOut[i] = w;
    T *= (1.0f - alpha + eps);
  }
  const float inv_eps = 1e10f;
  float disp = acc / depth;
  disp = (disp > 0.0f && disp < inv_eps && acc > 0.0f) ? disp : inv_eps;
  oRgb[r * 3 + 0] = cr + (1.0f - acc);
  oRgb[r * 3 + 1] = cg + (1.0f - acc);
  oRgb[r * 3 + 2] = cb + (1.0f - acc);
  oDisp[r] = disp;
  oAcc[r] = acc;
}

// inverse-CDF resampling + merge with coarse z's -> 192 sorted fine z's / ray
__global__ void pdf_resample(const float* __restrict__ zc,  // [RAYS][64]
                             const float* __restrict__ wc,  // [RAYS][64]
                             float* __restrict__ zf) {      // [RAYS][192]
  int r = blockIdx.x * blockDim.x + threadIdx.x;
  if (r >= RAYS) return;
  const float* z = zc + r * 64;
  float bins[63], cdf[63], w[62];
  float wsum = 0.0f;
  for (int i = 0; i < 62; ++i) { w[i] = wc[r * 64 + 1 + i]; wsum += w[i]; }
  for (int i = 0; i < 63; ++i) bins[i] = 0.5f * (z[i] + z[i + 1]);
  float pad = 1e-5f - wsum; pad = pad > 0.0f ? pad : 0.0f;
  float wpad = pad * (1.0f / 62.0f);
  float inv = 1.0f / (wsum + pad);
  cdf[0] = 0.0f;
  float run = 0.0f;
  for (int i = 1; i < 62; ++i) {
    run += (w[i - 1] + wpad) * inv;
    cdf[i] = run < 1.0f ? run : 1.0f;
  }
  cdf[62] = 1.0f;
  float zs[128];
  const float UMAX = 1.0f - 1.1920929e-07f;
  for (int j = 0; j < 128; ++j) {
    float u = (float)j * (UMAX / 127.0f);
    int i0 = 0, i1 = 62;
    for (int i = 0; i < 63; ++i) if (u >= cdf[i]) i0 = i;
    for (int i = 62; i >= 0; --i) if (cdf[i] > u) i1 = i;
    float c0 = cdf[i0], c1 = cdf[i1];
    float den = c1 - c0;
    float t = den > 0.0f ? (u - c0) / den : 0.0f;
    t = t < 0.0f ? 0.0f : (t > 1.0f ? 1.0f : t);
    zs[j] = bins[i0] + t * (bins[i1] - bins[i0]);
  }
  // merge 64 sorted coarse z's with 128 sorted samples
  int a = 0, b = 0;
  float* out = zf + (long)r * SF;
  for (int k = 0; k < SF; ++k) {
    float va = a < 64 ? z[a] : 3.0e38f;
    float vb = b < 128 ? zs[b] : 3.0e38f;
    if (va <= vb) { out[k] = va; ++a; } else { out[k] = vb; ++b; }
  }
}

// ---------------- host side ------------------------------------------------

extern "C" void kernel_launch(void* const* d_in, const int* in_sizes, int n_in,
                              void* d_out, int out_size, void* d_ws, size_t ws_size,
                              hipStream_t stream) {
  (void)in_sizes; (void)n_in; (void)out_size; (void)ws_size;
  const float* origins    = (const float*)d_in[0];
  const float* directions = (const float*)d_in[1];
  const float* viewdirs   = (const float*)d_in[2];
  // Param dicts assumed flattened pytree-style (alphabetical key order):
  // b0..b7, b_bneck, b_cond0, b_rgb, b_sigma, w0..w7, w_bneck, w_cond0, w_rgb, w_sigma
  // params0 at d_in[3..26], params1 at d_in[27..50].

  char* ws = (char*)d_ws;
  size_t off = 0;
  auto alloc = [&](size_t bytes) -> char* {
    char* p = ws + off;
    off = (off + bytes + 255) & ~(size_t)255;
    return p;
  };
  const long WT_HALVES = 600064;
  _Float16* wt0 = (_Float16*)alloc(WT_HALVES * 2);
  _Float16* wt1 = (_Float16*)alloc(WT_HALVES * 2);
  _Float16* vde = (_Float16*)alloc((size_t)RAYS * 32 * 2);
  _Float16* encC = (_Float16*)alloc((size_t)RAYS * SC * 64 * 2);
  _Float16* encF = (_Float16*)alloc((size_t)RAYS * SF * 64 * 2);
  float* zC   = (float*)alloc((size_t)RAYS * SC * 4);
  float* zF   = (float*)alloc((size_t)RAYS * SF * 4);
  float* sigC = (float*)alloc((size_t)RAYS * SC * 4);
  float* rgbC = (float*)alloc((size_t)RAYS * SC * 3 * 4);
  float* sigF = (float*)alloc((size_t)RAYS * SF * 4);
  float* rgbF = (float*)alloc((size_t)RAYS * SF * 3 * 4);
  float* wC   = (float*)alloc((size_t)RAYS * SC * 4);

  // weight conversion table: {src offset within params, fi, fo, Kp, Np, dstOff}
  struct Cvt { int si, fi, fo, Kp, Np; long dst; };
  static const Cvt cv[12] = {
      {12,  63, 256,  64, 256,      0},  // w0
      {13, 256, 256, 256, 256,  16384},  // w1
      {14, 256, 256, 256, 256,  81920},  // w2
      {15, 256, 256, 256, 256, 147456},  // w3
      {16, 256, 256, 256, 256, 212992},  // w4
      {17, 319, 256, 320, 256, 278528},  // w5 (skip layer)
      {18, 256, 256, 256, 256, 360448},  // w6
      {19, 256, 256, 256, 256, 425984},  // w7
      {23, 256,   1, 256,  16, 491520},  // w_sigma
      {20, 256, 256, 256, 256, 495616},  // w_bneck
      {21, 283, 128, 288, 128, 561152},  // w_cond0
      {22, 128,   3, 128,  16, 598016},  // w_rgb
  };
  _Float16* wts[2] = {wt0, wt1};
  for (int mlp = 0; mlp < 2; ++mlp) {
    int pb = 3 + mlp * 24;
    for (int i = 0; i < 12; ++i) {
      const Cvt& c = cv[i];
      int n = c.Kp * c.Np;
      convert_wt<<<(n + 255) / 256, 256, 0, stream>>>(
          (const float*)d_in[pb + c.si], c.fi, c.fo, c.Kp, c.Np, wts[mlp] + c.dst);
    }
  }

  auto makeW = [&](int mlp) -> MlpW {
    int pb = 3 + mlp * 24;
    _Float16* base = wts[mlp];
    MlpW W;
    W.w0 = base + 0;      W.w1 = base + 16384;  W.w2 = base + 81920;
    W.w3 = base + 147456; W.w4 = base + 212992; W.w5 = base + 278528;
    W.w6 = base + 360448; W.w7 = base + 425984;
    W.wsig = base + 491520; W.wbneck = base + 495616;
    W.wcond = base + 561152; W.wrgb = base + 598016;
    W.b0 = (const float*)d_in[pb + 0]; W.b1 = (const float*)d_in[pb + 1];
    W.b2 = (const float*)d_in[pb + 2]; W.b3 = (const float*)d_in[pb + 3];
    W.b4 = (const float*)d_in[pb + 4]; W.b5 = (const float*)d_in[pb + 5];
    W.b6 = (const float*)d_in[pb + 6]; W.b7 = (const float*)d_in[pb + 7];
    W.bbneck = (const float*)d_in[pb + 8];  W.bcond = (const float*)d_in[pb + 9];
    W.brgb   = (const float*)d_in[pb + 10]; W.bsig  = (const float*)d_in[pb + 11];
    return W;
  };
  MlpW W0 = makeW(0), W1 = makeW(1);

  vd_enc<<<(RAYS + 255) / 256, 256, 0, stream>>>(viewdirs, vde);
  coarse_enc<<<(RAYS * SC + 255) / 256, 256, 0, stream>>>(origins, directions, encC, zC);

  const size_t shmem = (size_t)(2 * TM * LDST + TM * 64) * sizeof(_Float16); // 184320 B
  (void)hipFuncSetAttribute(reinterpret_cast<const void*>(&nerf_mlp),
                            hipFuncAttributeMaxDynamicSharedMemorySize, (int)shmem);

  float* out = (float*)d_out;
  nerf_mlp<<<(RAYS * SC) / TM, 256, shmem, stream>>>(encC, vde, W0, sigC, rgbC, SC);
  render<<<(RAYS + 255) / 256, 256, 0, stream>>>(rgbC, sigC, zC, directions,
                                                 out + 0, out + 3072, out + 4096, wC, SC);
  pdf_resample<<<(RAYS + 255) / 256, 256, 0, stream>>>(zC, wC, zF);
  fine_enc<<<(RAYS * SF + 255) / 256, 256, 0, stream>>>(origins, directions, zF, encF);
  nerf_mlp<<<(RAYS * SF) / TM, 256, shmem, stream>>>(encF, vde, W1, sigF, rgbF, SF);
  render<<<(RAYS + 255) / 256, 256, 0, stream>>>(rgbF, sigF, zF, directions,
                                                 out + 5120, out + 8192, out + 9216,
                                                 (float*)nullptr, SF);
}
